// OptimizedNeuralODE_10033043604047
// MI455X (gfx1250) — matrix operand order repaired and checked by reading, not compiled, not run on hardware
//
#include <hip/hip_runtime.h>

#define B_ 8
#define S_ 64
#define N_ 1024
#define C_ 16
#define H_ 128
#define P_ 10

typedef __attribute__((ext_vector_type(16))) __bf16 bf16x16;
typedef __attribute__((ext_vector_type(8))) float f32x8;

__device__ __forceinline__ __bf16 f2bf(float x) { return (__bf16)x; }   // native v_cvt (RNE)

__device__ __forceinline__ float fixnum(float v) {
  const float M = 3.4028234663852886e38f;
  if (v != v) return 0.0f;
  if (v > M) return M;
  if (v < -M) return -M;
  return v;
}

__device__ __forceinline__ float silu_f(float v) { return v / (1.0f + __expf(-v)); }

// ---- WMMA fragment loaders (CDNA5 16x16x32 bf16 layouts, ISA 7.12.2) ----
// A (16x32, MxK): lane L, M = L&15; elems 0..7 -> K = koff+e, elems 8..15 -> K = koff+16+e, koff=(L>>4)*8
__device__ __forceinline__ bf16x16 fragA_f32(const float* base, int ld, int row0, int k0, int lane) {
  const float* p = base + (size_t)(row0 + (lane & 15)) * ld + k0 + ((lane >> 4) << 3);
  bf16x16 f;
#pragma unroll
  for (int e = 0; e < 8; ++e) f[e] = f2bf(p[e]);
#pragma unroll
  for (int e = 0; e < 8; ++e) f[8 + e] = f2bf(p[16 + e]);
  return f;
}

__device__ __forceinline__ bf16x16 fragA_bf(const __bf16* base, int ld, int row0, int k0, int lane) {
  const __bf16* p = base + (size_t)(row0 + (lane & 15)) * ld + k0 + ((lane >> 4) << 3);
  bf16x16 f;
#pragma unroll
  for (int e = 0; e < 8; ++e) f[e] = p[e];
#pragma unroll
  for (int e = 0; e < 8; ++e) f[8 + e] = p[16 + e];
  return f;
}

// B (32x16, KxN) from transposed storage T[col][K]: lane L, N = L&15; elems 0..7 -> K = koff+e, elems 8..15 -> K = koff+16+e, koff=(L>>4)*8
__device__ __forceinline__ bf16x16 fragB_bfT(const __bf16* baseT, int ld, int col0, int k0, int lane) {
  const __bf16* p = baseT + (size_t)(col0 + (lane & 15)) * ld + k0 + ((lane >> 4) << 3);
  bf16x16 f;
#pragma unroll
  for (int e = 0; e < 8; ++e) f[e] = p[e];
#pragma unroll
  for (int e = 0; e < 8; ++e) f[8 + e] = p[16 + e];
  return f;
}

__device__ __forceinline__ f32x8 wmma_bf16(bf16x16 a, bf16x16 b, f32x8 c) {
  return __builtin_amdgcn_wmma_f32_16x16x32_bf16(false, a, false, b, (short)0, c, false, false);
}

// Async global -> LDS 16-byte copy (gfx1250; tracked by ASYNCcnt).
// Low 32 bits of a generic shared pointer are the LDS byte address (ISA 10.2 aperture map).
__device__ __forceinline__ void async_copy_b128(void* lds_ptr, const void* gptr) {
  unsigned la = (unsigned)(size_t)lds_ptr;
  asm volatile("global_load_async_to_lds_b128 %0, %1, off" :: "v"(la), "v"(gptr) : "memory");
}
__device__ __forceinline__ void wait_async0() {
  asm volatile("s_wait_asynccnt 0" ::: "memory");
}

// ---- prep kernels ----
__global__ void k_transpose_bf(const float* __restrict__ src, __bf16* __restrict__ dst,
                               int K, int Nc) {
  int i = blockIdx.x * blockDim.x + threadIdx.x;
  if (i >= K * Nc) return;
  int k = i / Nc, n = i - k * Nc;
  dst[(size_t)n * K + k] = f2bf(src[i]);
}

__global__ void k_norm_adj(const float* __restrict__ adj, __bf16* __restrict__ Abf) {
  __shared__ float red[256];
  int m = blockIdx.x, t = threadIdx.x;
  const float* row = adj + (size_t)m * N_;
  float s = 0.f;
  for (int n = t; n < N_; n += 256) s += row[n];
  red[t] = s; __syncthreads();
  for (int off = 128; off > 0; off >>= 1) { if (t < off) red[t] += red[t + off]; __syncthreads(); }
  float inv = 1.0f / (red[0] + 1e-8f);
  for (int n = t; n < N_; n += 256) Abf[(size_t)m * N_ + n] = f2bf(row[n] * inv);
}

// ---- encoder: z0 = silu(LN(x[:,0] @ W + b)); block = one (b,n) row, 128 threads ----
__global__ void k_encoder(const float* __restrict__ x, const float* __restrict__ w,
                          const float* __restrict__ bias, const float* __restrict__ g,
                          const float* __restrict__ beta, float* __restrict__ s,
                          __bf16* __restrict__ sbf, float* __restrict__ sol0) {
  __shared__ float xr[C_];
  __shared__ float red[H_];
  int R = blockIdx.x;            // b*N + n
  int h = threadIdx.x;
  int b = R >> 10, n = R & (N_ - 1);
  const float* xp = x + ((size_t)(b * S_) * N_ + n) * C_;   // x[b,0,n,:]
  if (h < C_) xr[h] = xp[h];
  __syncthreads();
  float acc = bias[h];
#pragma unroll
  for (int k = 0; k < C_; ++k) acc += xr[k] * w[k * H_ + h];
  red[h] = acc; __syncthreads();
  for (int off = 64; off > 0; off >>= 1) { if (h < off) red[h] += red[h + off]; __syncthreads(); }
  float mean = red[0] * (1.0f / H_); __syncthreads();
  float d = acc - mean;
  red[h] = d * d; __syncthreads();
  for (int off = 64; off > 0; off >>= 1) { if (h < off) red[h] += red[h + off]; __syncthreads(); }
  float var = red[0] * (1.0f / H_);
  float v = silu_f(d * rsqrtf(var + 1e-5f) * g[h] + beta[h]);
  size_t o = (size_t)R * H_ + h;
  s[o] = v; sbf[o] = f2bf(v); sol0[o] = v;
}

// ---- attention: softmax over the 1024 nodes per batch (block = batch, 1024 threads) ----
__global__ void k_attn(const float* __restrict__ s, const float* __restrict__ aw,
                       const float* __restrict__ ab, float* __restrict__ attnw) {
  __shared__ float red[N_];
  int b = blockIdx.x, n = threadIdx.x;
  const float* sp = s + ((size_t)b * N_ + n) * H_;
  float l = ab[0];
#pragma unroll 8
  for (int h = 0; h < H_; ++h) l += sp[h] * aw[h];
  red[n] = l; __syncthreads();
  for (int off = 512; off > 0; off >>= 1) { if (n < off) red[n] = fmaxf(red[n], red[n + off]); __syncthreads(); }
  float mx = red[0]; __syncthreads();
  float e = __expf(l - mx);
  red[n] = e; __syncthreads();
  for (int off = 512; off > 0; off >>= 1) { if (n < off) red[n] += red[n + off]; __syncthreads(); }
  attnw[(size_t)b * N_ + n] = e / red[0];
}

// ---- SWT[b][h][n] = bf16(nan_to_num(s[b][n][h]) * attn[b][n])  (transposed for WMMA-B) ----
__global__ void k_build_swt(const float* __restrict__ s, const float* __restrict__ attnw,
                            __bf16* __restrict__ swt) {
  int i = blockIdx.x * blockDim.x + threadIdx.x;
  if (i >= B_ * N_ * H_) return;
  int h = i & (H_ - 1);
  int n = (i >> 7) & (N_ - 1);
  int b = i >> 17;
  float v = fixnum(s[i]) * attnw[(b << 10) + n];
  swt[((size_t)b * H_ + h) * N_ + n] = f2bf(v);
}

// ---- dynamics layer 1: tmp1 = bf16(silu(LN(s @ W1 + b1)))  [WMMA, 16-row tile per block] ----
__global__ void k_dyn1(const __bf16* __restrict__ sbf, const __bf16* __restrict__ WT,
                       const float* __restrict__ bias, const float* __restrict__ g,
                       const float* __restrict__ beta, __bf16* __restrict__ out) {
  __shared__ float tile[16][H_ + 8];
  __shared__ float pr[16][16], pr2[16][16];
  __shared__ float mean_s[16], inv_s[16];
  int wave = threadIdx.x >> 5, lane = threadIdx.x & 31;
  int row0 = blockIdx.x << 4;
  int col0 = wave << 4;
  f32x8 acc = {0.f, 0.f, 0.f, 0.f, 0.f, 0.f, 0.f, 0.f};
#pragma unroll
  for (int k = 0; k < H_; k += 32)
    acc = wmma_bf16(fragA_bf(sbf, H_, row0, k, lane), fragB_bfT(WT, H_, col0, k, lane), acc);
  int n = (lane & 15) + col0;
  int mh = (lane >> 4) << 3;
  float bn = bias[n];
#pragma unroll
  for (int r = 0; r < 8; ++r) tile[mh + r][n] = acc[r] + bn;
  __syncthreads();
  int t = threadIdx.x, row = t >> 4, seg = t & 15;
  float ps = 0.f, pq = 0.f;
#pragma unroll
  for (int c = 0; c < 8; ++c) { float v = tile[row][seg * 8 + c]; ps += v; pq += v * v; }
  pr[row][seg] = ps; pr2[row][seg] = pq;
  __syncthreads();
  if (seg == 0) {
    float S1 = 0.f, S2 = 0.f;
#pragma unroll
    for (int j = 0; j < 16; ++j) { S1 += pr[row][j]; S2 += pr2[row][j]; }
    float m = S1 * (1.0f / H_);
    float var = S2 * (1.0f / H_) - m * m;
    mean_s[row] = m; inv_s[row] = rsqrtf(var + 1e-5f);
  }
  __syncthreads();
  float m = mean_s[row], iv = inv_s[row];
  size_t ro = (size_t)(row0 + row) * H_;
#pragma unroll
  for (int c = 0; c < 8; ++c) {
    int cc = seg * 8 + c;
    float v = (tile[row][cc] - m) * iv * g[cc] + beta[cc];
    out[ro + cc] = f2bf(silu_f(v));
  }
}

// ---- dynamics layer 2: dyn = tanh(tmp1 @ W2 + b2) ----
__global__ void k_dyn2(const __bf16* __restrict__ tmp1, const __bf16* __restrict__ WT,
                       const float* __restrict__ bias, float* __restrict__ dyn) {
  int wave = threadIdx.x >> 5, lane = threadIdx.x & 31;
  int row0 = blockIdx.x << 4;
  int col0 = wave << 4;
  f32x8 acc = {0.f, 0.f, 0.f, 0.f, 0.f, 0.f, 0.f, 0.f};
#pragma unroll
  for (int k = 0; k < H_; k += 32)
    acc = wmma_bf16(fragA_bf(tmp1, H_, row0, k, lane), fragB_bfT(WT, H_, col0, k, lane), acc);
  int n = (lane & 15) + col0;
  int mh = (lane >> 4) << 3;
  float bn = bias[n];
#pragma unroll
  for (int r = 0; r < 8; ++r)
    dyn[(size_t)(row0 + mh + r) * H_ + n] = tanhf(acc[r] + bn);
}

// ---- diffusion GEMM + norm clamp + Euler update: block = (m-tile, batch) ----
// The A tile is identical for all 8 waves of the block, so it is staged into
// LDS in 16x128 (4KB) chunks with async global->LDS b128 copies: all 256
// threads copy one 16B chunk, double buffered, so the copy of chunk c+1
// overlaps an unrolled group of 4 WMMAs + B-fragment loads on chunk c.
// 8 block barriers per 1024-deep K loop instead of 32.
__global__ void k_diffuse(const __bf16* __restrict__ Abf, const __bf16* __restrict__ swt,
                          const float* __restrict__ dyn, const float* __restrict__ dscale,
                          const float* __restrict__ tscale, float* __restrict__ s,
                          __bf16* __restrict__ sbf, float* __restrict__ solp, float dt) {
  __shared__ __bf16 atile[2][16][H_];     // 2 x 4KB double buffer
  __shared__ float ssq[16];
  int tid = threadIdx.x;
  int wave = tid >> 5, lane = tid & 31;
  int row0 = blockIdx.x << 4;          // node tile (64 tiles)
  int b = blockIdx.y;                  // batch
  int col0 = wave << 4;                // H tile
  const __bf16* swtb = swt + (size_t)b * H_ * N_;

  // each thread stages one 16B chunk of the 16x128 A chunk
  int arow = tid >> 4, acol = (tid & 15) << 3;
  const __bf16* agp = Abf + (size_t)(row0 + arow) * N_ + acol;

  async_copy_b128(&atile[0][arow][acol], agp);
  wait_async0();
  __syncthreads();

  f32x8 acc = {0.f, 0.f, 0.f, 0.f, 0.f, 0.f, 0.f, 0.f};
  for (int c = 0; c < 8; ++c) {                       // 8 chunks of K=128
    int cur = c & 1;
    if (c + 1 < 8)
      async_copy_b128(&atile[cur ^ 1][arow][acol], agp + (c + 1) * H_);
    __builtin_prefetch(swtb + (size_t)(col0 + (lane & 15)) * N_ + (c + 1) * H_, 0, 1);
#pragma unroll
    for (int kk = 0; kk < 4; ++kk)
      acc = wmma_bf16(fragA_bf(&atile[cur][0][0], H_, 0, kk * 32, lane),
                      fragB_bfT(swtb, N_, col0, c * H_ + kk * 32, lane), acc);
    wait_async0();
    __syncthreads();
  }

  if (tid < 16) ssq[tid] = 0.f;
  __syncthreads();
  int n = (lane & 15) + col0;
  int mh = (lane >> 4) << 3;
  float ds = dscale[0], ts = tscale[0];
  float dx[8];
#pragma unroll
  for (int r = 0; r < 8; ++r) {
    int m = mh + r;
    size_t off = ((size_t)(b * N_) + row0 + m) * H_ + n;
    float v = ts * (dyn[off] + acc[r] * ds);
    dx[r] = v;
    atomicAdd(&ssq[m], v * v);
  }
  __syncthreads();
#pragma unroll
  for (int r = 0; r < 8; ++r) {
    int m = mh + r;
    float sc = fminf(10.0f / (sqrtf(ssq[m]) + 1e-8f), 1.0f);
    size_t off = ((size_t)(b * N_) + row0 + m) * H_ + n;
    float v = fixnum(dx[r] * sc);
    float sn = s[off] + v * dt;
    s[off] = sn;
    sbf[off] = f2bf(sn);
    solp[off] = sn;
  }
}

// ---- linear interpolation P=10 grid -> S=64 grid (writes output #2) ----
__global__ void k_interp(const float* __restrict__ sol, float* __restrict__ out2) {
  const int total = S_ * B_ * N_ * H_;
  for (int i = blockIdx.x * blockDim.x + threadIdx.x; i < total; i += gridDim.x * blockDim.x) {
    int st = i >> 20;                       // B*N*H = 1<<20
    int rem = i & ((1 << 20) - 1);
    float v;
    if (st == 0) {
      v = sol[rem];
    } else {
      int ic = (st + 6) / 7;                // ceil(st/7), in [1,9]
      float alpha = (float)(st - 7 * (ic - 1)) * (1.0f / 7.0f);
      float a0 = sol[(size_t)(ic - 1) * (1 << 20) + rem];
      float a1 = sol[(size_t)ic * (1 << 20) + rem];
      v = (1.0f - alpha) * a0 + alpha * a1;
    }
    out2[i] = v;
  }
}

// ---- decoder, fully fused: pred = (silu(LN(interp@W1+b1)))@W2 + b2 ----
__global__ void k_decoder(const float* __restrict__ interp, const __bf16* __restrict__ W1T,
                          const float* __restrict__ b1, const float* __restrict__ g,
                          const float* __restrict__ beta, const __bf16* __restrict__ W2T,
                          const float* __restrict__ b2, float* __restrict__ pred) {
  __shared__ float tile[16][H_ + 8];
  __shared__ float pr[16][16], pr2[16][16];
  __shared__ float mean_s[16], inv_s[16];
  int wave = threadIdx.x >> 5, lane = threadIdx.x & 31;
  int row0 = blockIdx.x << 4;
  int col0 = wave << 4;
  f32x8 acc = {0.f, 0.f, 0.f, 0.f, 0.f, 0.f, 0.f, 0.f};
#pragma unroll
  for (int k = 0; k < H_; k += 32)
    acc = wmma_bf16(fragA_f32(interp, H_, row0, k, lane), fragB_bfT(W1T, H_, col0, k, lane), acc);
  int n = (lane & 15) + col0;
  int mh = (lane >> 4) << 3;
  float bn = b1[n];
#pragma unroll
  for (int r = 0; r < 8; ++r) tile[mh + r][n] = acc[r] + bn;
  __syncthreads();
  int t = threadIdx.x, row = t >> 4, seg = t & 15;
  float ps = 0.f, pq = 0.f;
#pragma unroll
  for (int c = 0; c < 8; ++c) { float v = tile[row][seg * 8 + c]; ps += v; pq += v * v; }
  pr[row][seg] = ps; pr2[row][seg] = pq;
  __syncthreads();
  if (seg == 0) {
    float S1 = 0.f, S2 = 0.f;
#pragma unroll
    for (int j = 0; j < 16; ++j) { S1 += pr[row][j]; S2 += pr2[row][j]; }
    float m = S1 * (1.0f / H_);
    float var = S2 * (1.0f / H_) - m * m;
    mean_s[row] = m; inv_s[row] = rsqrtf(var + 1e-5f);
  }
  __syncthreads();
  {
    float m = mean_s[row], iv = inv_s[row];
#pragma unroll
    for (int c = 0; c < 8; ++c) {            // each thread rewrites only its own 8 cells
      int cc = seg * 8 + c;
      tile[row][cc] = silu_f((tile[row][cc] - m) * iv * g[cc] + beta[cc]);
    }
  }
  __syncthreads();
  if (wave == 0) {                           // wave-uniform branch: EXEC all-1s for WMMA
    f32x8 acc2 = {0.f, 0.f, 0.f, 0.f, 0.f, 0.f, 0.f, 0.f};
#pragma unroll
    for (int k = 0; k < H_; k += 32)
      acc2 = wmma_bf16(fragA_f32(&tile[0][0], H_ + 8, 0, k, lane), fragB_bfT(W2T, H_, 0, k, lane), acc2);
    int c = lane & 15;
    int mh0 = (lane >> 4) << 3;
    float bc = b2[c];
#pragma unroll
    for (int r = 0; r < 8; ++r) {
      int R = row0 + mh0 + r;                // R = ((st*B + b)*N + n)
      int nn = R & (N_ - 1);
      int bb = (R >> 10) & 7;
      int st = R >> 13;
      pred[(((size_t)(bb * S_ + st)) * N_ + nn) * C_ + c] = acc2[r] + bc;
    }
  }
}

extern "C" void kernel_launch(void* const* d_in, const int* in_sizes, int n_in,
                              void* d_out, int out_size, void* d_ws, size_t ws_size,
                              hipStream_t stream) {
  (void)in_sizes; (void)n_in; (void)out_size; (void)ws_size;
  const float* x          = (const float*)d_in[0];
  const float* adj        = (const float*)d_in[1];
  const float* enc_w      = (const float*)d_in[2];
  const float* enc_b      = (const float*)d_in[3];
  const float* enc_g      = (const float*)d_in[4];
  const float* enc_beta   = (const float*)d_in[5];
  const float* dyn_w1     = (const float*)d_in[6];
  const float* dyn_b1     = (const float*)d_in[7];
  const float* dyn_g      = (const float*)d_in[8];
  const float* dyn_beta   = (const float*)d_in[9];
  const float* dyn_w2     = (const float*)d_in[10];
  const float* dyn_b2     = (const float*)d_in[11];
  const float* attn_w     = (const float*)d_in[12];
  const float* attn_b     = (const float*)d_in[13];
  const float* diff_scale = (const float*)d_in[14];
  const float* time_scale = (const float*)d_in[15];
  const float* dec_w1     = (const float*)d_in[16];
  const float* dec_b1     = (const float*)d_in[17];
  const float* dec_g      = (const float*)d_in[18];
  const float* dec_beta   = (const float*)d_in[19];
  const float* dec_w2     = (const float*)d_in[20];
  const float* dec_b2     = (const float*)d_in[21];

  float* out1 = (float*)d_out;                                     // pred [B,S,N,C]
  float* out2 = out1 + (size_t)B_ * S_ * N_ * C_;                  // interp [S,B,N,H]

  // workspace carve-out (~56 MB total)
  char* w = (char*)d_ws;
  auto alloc = [&](size_t bytes) -> void* {
    void* p = (void*)w; w += (bytes + 255) & ~(size_t)255; return p;
  };
  __bf16* Abf  = (__bf16*)alloc((size_t)N_ * N_ * 2);
  __bf16* WT1  = (__bf16*)alloc((size_t)H_ * H_ * 2);
  __bf16* WT2  = (__bf16*)alloc((size_t)H_ * H_ * 2);
  __bf16* WD1T = (__bf16*)alloc((size_t)H_ * H_ * 2);
  __bf16* WD2T = (__bf16*)alloc((size_t)C_ * H_ * 2);
  float*  st   = (float*)alloc((size_t)B_ * N_ * H_ * 4);
  __bf16* stbf = (__bf16*)alloc((size_t)B_ * N_ * H_ * 2);
  __bf16* tmp1 = (__bf16*)alloc((size_t)B_ * N_ * H_ * 2);
  float*  dyn  = (float*)alloc((size_t)B_ * N_ * H_ * 4);
  float*  attw = (float*)alloc((size_t)B_ * N_ * 4);
  __bf16* swt  = (__bf16*)alloc((size_t)B_ * H_ * N_ * 2);
  float*  sol  = (float*)alloc((size_t)P_ * B_ * N_ * H_ * 4);

  // prep: weights -> bf16 transposed; adjacency -> row-normalized bf16
  k_transpose_bf<<<(H_ * H_ + 255) / 256, 256, 0, stream>>>(dyn_w1, WT1, H_, H_);
  k_transpose_bf<<<(H_ * H_ + 255) / 256, 256, 0, stream>>>(dyn_w2, WT2, H_, H_);
  k_transpose_bf<<<(H_ * H_ + 255) / 256, 256, 0, stream>>>(dec_w1, WD1T, H_, H_);
  k_transpose_bf<<<(H_ * C_ + 255) / 256, 256, 0, stream>>>(dec_w2, WD2T, H_, C_);
  k_norm_adj<<<N_, 256, 0, stream>>>(adj, Abf);

  // encoder -> state (fp32 + bf16) + sol[0]
  k_encoder<<<B_ * N_, H_, 0, stream>>>(x, enc_w, enc_b, enc_g, enc_beta, st, stbf, sol);

  // 9 Euler steps, dt = 63/9 = 7
  for (int p = 1; p < P_; ++p) {
    k_attn<<<B_, N_, 0, stream>>>(st, attn_w, attn_b, attw);
    k_build_swt<<<(B_ * N_ * H_ + 255) / 256, 256, 0, stream>>>(st, attw, swt);
    k_dyn1<<<(B_ * N_) / 16, 256, 0, stream>>>(stbf, WT1, dyn_b1, dyn_g, dyn_beta, tmp1);
    k_dyn2<<<(B_ * N_) / 16, 256, 0, stream>>>(tmp1, WT2, dyn_b2, dyn);
    dim3 gdiff(N_ / 16, B_);
    k_diffuse<<<gdiff, 256, 0, stream>>>(Abf, swt, dyn, diff_scale, time_scale, st, stbf,
                                         sol + (size_t)p * B_ * N_ * H_, 7.0f);
  }

  // interpolate to the full sequence grid (output #2), then fused decoder (output #1)
  k_interp<<<16384, 256, 0, stream>>>(sol, out2);
  k_decoder<<<(S_ * B_ * N_) / 16, 256, 0, stream>>>(out2, WD1T, dec_b1, dec_g, dec_beta,
                                                     WD2T, dec_b2, out1);
}
